// ModuleGATInteraction_9122510537163
// MI455X (gfx1250) — matrix operand order repaired and checked by reading, not compiled, not run on hardware
//
#include <hip/hip_runtime.h>
#include <hip/hip_bf16.h>
#include <stdint.h>

typedef __attribute__((ext_vector_type(16))) _Float16 v16h;
typedef __attribute__((ext_vector_type(8)))  _Float16 v8h;
typedef __attribute__((ext_vector_type(8)))  float    v8f;

#define LRELU(v, s) ((v) >= 0.0f ? (v) : (s) * (v))

// Order-preserving float -> uint map so we can use u32 atomicMax for segment-max.
__device__ __forceinline__ unsigned orderF(float f) {
  unsigned u = __float_as_uint(f);
  return (u & 0x80000000u) ? ~u : (u | 0x80000000u);
}
__device__ __forceinline__ float unorderF(unsigned u) {
  return (u & 0x80000000u) ? __uint_as_float(u & 0x7FFFFFFFu) : __uint_as_float(~u);
}

// ---------------- utility kernels ----------------

__global__ void k_zero(uint32_t* __restrict__ p, size_t count) {
  size_t i = (size_t)blockIdx.x * blockDim.x + threadIdx.x;
  size_t stride = (size_t)gridDim.x * blockDim.x;
  for (; i < count; i += stride) p[i] = 0u;
}

__global__ void k_f32_to_f16(const float* __restrict__ src, _Float16* __restrict__ dst,
                             size_t count) {
  size_t i = (size_t)blockIdx.x * blockDim.x + threadIdx.x;
  size_t stride = (size_t)gridDim.x * blockDim.x;
  for (; i < count; i += stride) dst[i] = (_Float16)src[i];
}

// W: [F, 256] f32  ->  WT: [256, F] f16 (so the WMMA B-fragment loads are contiguous)
__global__ void k_transpose_f16(const float* __restrict__ W, _Float16* __restrict__ WT, int F) {
  int i = blockIdx.x * blockDim.x + threadIdx.x;
  int total = 256 * F;
  if (i >= total) return;
  int c = i / F, f = i - c * F;
  WT[i] = (_Float16)W[(size_t)f * 256 + c];
}

// Fold att_src/att_dst into A:  Acomb[f][j] (j=0..3: src heads, j=4..7: dst heads)
__global__ void k_combine_att(const float* __restrict__ A, const float* __restrict__ att_src,
                              const float* __restrict__ att_dst, float* __restrict__ Acomb,
                              int F) {
  int i = blockIdx.x * blockDim.x + threadIdx.x;
  if (i >= F * 8) return;
  int f = i >> 3, j = i & 7;
  int h = j & 3;
  const float* att = (j & 4) ? att_dst : att_src;
  float s = 0.f;
  for (int d = 0; d < 4; ++d) s += A[(size_t)f * 16 + h * 4 + d] * att[h * 4 + d];
  Acomb[i] = s;
}

// alpha[n][0..3]=src scores, alpha[n][4..7]=dst scores  (tiny GEMM, f32 VALU)
__global__ void k_alpha(const float* __restrict__ x, const float* __restrict__ Acomb,
                        float* __restrict__ alpha, int n, int F) {
  int i = blockIdx.x * blockDim.x + threadIdx.x;
  if (i >= n * 8) return;
  int nn = i >> 3, j = i & 7;
  const float* xr = x + (size_t)nn * F;
  float s = 0.f;
  for (int f = 0; f < F; ++f) s += xr[f] * Acomb[f * 8 + j];
  alpha[i] = s;
}

// ---------------- WMMA node-feature GEMM: H = Xh(f16) @ W(f16) -> f32 [n,256] ----------------
// One wave computes a 16x16 output tile. 4 waves / block cover 4 column tiles.

__global__ void __launch_bounds__(128) k_wmma_gemm(const _Float16* __restrict__ Xh,  // [n, F]
                                                   const _Float16* __restrict__ WT,  // [256, F]
                                                   float* __restrict__ Hout,         // [n, 256]
                                                   int n, int F) {
  const int lane = threadIdx.x & 31;
  const int waveInBlock = threadIdx.x >> 5;
  const int rowBase = blockIdx.x * 16;
  const int colBase = (blockIdx.y * 4 + waveInBlock) * 16;
  const int lr = lane & 15;
  const int hs = lane >> 4;  // half-wave select

  int row = rowBase + lr;
  int rowL = (row < n) ? row : (n - 1);  // clamp to avoid OOB loads on the tail tile

  const int kbA = hs * 8;    // A: lanes 0-15 -> K 0-7 / 16-23 ; lanes 16-31 -> K 8-15 / 24-31
  const int kbB = hs * 8;    // B: lanes 0-15 -> K 0-7 / 16-23 ; lanes 16-31 -> K 8-15 / 24-31

  v8f c = {};
  for (int k0 = 0; k0 < F; k0 += 32) {
    const _Float16* aptr = Xh + (size_t)rowL * F + k0 + kbA;
    v8h alo = *(const v8h*)(aptr);
    v8h ahi = *(const v8h*)(aptr + 16);
    v16h a;
    for (int j = 0; j < 8; ++j) { a[j] = alo[j]; a[8 + j] = ahi[j]; }

    const _Float16* bptr = WT + (size_t)(colBase + lr) * F + k0 + kbB;
    v8h blo = *(const v8h*)(bptr);
    v8h bhi = *(const v8h*)(bptr + 16);
    v16h b;
    for (int j = 0; j < 8; ++j) { b[j] = blo[j]; b[8 + j] = bhi[j]; }

    c = __builtin_amdgcn_wmma_f32_16x16x32_f16(false, a, false, b, (short)0, c, false, false);
  }

  // C/D layout: VGPR r -> M = r (lanes 0-15) / 8+r (lanes 16-31), N = lane%16
  const int mBase = rowBase + hs * 8;
  for (int r = 0; r < 8; ++r) {
    int m = mBase + r;
    if (m < n) Hout[(size_t)m * 256 + colBase + lr] = c[r];
  }
}

// ---------------- edge passes ----------------

__global__ void k_edge_max(const long long* __restrict__ ei, const float* __restrict__ alpha,
                           unsigned* __restrict__ m, long long E) {
  long long idx = (long long)blockIdx.x * blockDim.x + threadIdx.x;
  if (idx >= E * 4) return;
  long long e = idx >> 2;
  int h = (int)(idx & 3);
  int s = (int)ei[e], d = (int)ei[E + e];
  float v = alpha[(size_t)s * 8 + h] + alpha[(size_t)d * 8 + 4 + h];
  v = LRELU(v, 0.2f);
  atomicMax(m + (size_t)d * 4 + h, orderF(v));
}

__global__ void k_edge_denom(const long long* __restrict__ ei, const float* __restrict__ alpha,
                             const unsigned* __restrict__ m, float* __restrict__ den,
                             long long E) {
  long long idx = (long long)blockIdx.x * blockDim.x + threadIdx.x;
  if (idx >= E * 4) return;
  long long e = idx >> 2;
  int h = (int)(idx & 3);
  int s = (int)ei[e], d = (int)ei[E + e];
  float v = alpha[(size_t)s * 8 + h] + alpha[(size_t)d * 8 + 4 + h];
  v = LRELU(v, 0.2f);
  atomicAdd(den + (size_t)d * 4 + h, __expf(v - unorderF(m[(size_t)d * 4 + h])));
}

// One wave per edge; each lane carries 8 contiguous channels (one head per lane group of 8).
__global__ void k_edge_scatter(const long long* __restrict__ ei, const float* __restrict__ alpha,
                               const unsigned* __restrict__ m, const float* __restrict__ den,
                               const float* __restrict__ H, float* __restrict__ acc,
                               long long E) {
  const int lane = threadIdx.x & 31;
  const int h = lane >> 3;  // 8 lanes per head (8 ch * 8 lanes = 64 ch/head)
  long long wave = ((long long)blockIdx.x * blockDim.x + threadIdx.x) >> 5;
  long long nwaves = ((long long)gridDim.x * blockDim.x) >> 5;
  for (long long e = wave; e < E; e += nwaves) {
    int s = (int)ei[e], d = (int)ei[E + e];
    float v = alpha[(size_t)s * 8 + h] + alpha[(size_t)d * 8 + 4 + h];
    v = LRELU(v, 0.2f);
    float w = __expf(v - unorderF(m[(size_t)d * 4 + h])) / den[(size_t)d * 4 + h];
    const float4* hsrc = (const float4*)(H + (size_t)s * 256 + lane * 8);
    float4 v0 = hsrc[0], v1 = hsrc[1];
    float* op = acc + (size_t)d * 256 + lane * 8;
    atomicAdd(op + 0, v0.x * w); atomicAdd(op + 1, v0.y * w);
    atomicAdd(op + 2, v0.z * w); atomicAdd(op + 3, v0.w * w);
    atomicAdd(op + 4, v1.x * w); atomicAdd(op + 5, v1.y * w);
    atomicAdd(op + 6, v1.z * w); atomicAdd(op + 7, v1.w * w);
  }
}

// head-mean + bias + leaky_relu(0.1)
__global__ void k_finalize(const float* __restrict__ acc, const float* __restrict__ bias,
                           float* __restrict__ out, int n) {
  int i = blockIdx.x * blockDim.x + threadIdx.x;
  if (i >= n * 64) return;
  int nn = i >> 6, c = i & 63;
  const float* a = acc + (size_t)nn * 256;
  float s = 0.25f * (a[c] + a[64 + c] + a[128 + c] + a[192 + c]) + bias[c];
  out[i] = LRELU(s, 0.1f);
}

// ---------------- host-side layer driver ----------------

static void run_layer(const float* xin, int F, int n, long long E, const long long* ei,
                      const float* W, const float* A, const float* asrc, const float* adst,
                      const float* bias, float* out,
                      float* Hbuf, float* Acc, float* Alpha, unsigned* Mbuf, float* Den,
                      _Float16* Xh, _Float16* WT, float* Acomb, hipStream_t stream) {
  const int TB = 256;
  size_t cnt;

  cnt = (size_t)n * 256;
  k_zero<<<(unsigned)((cnt + TB - 1) / TB), TB, 0, stream>>>((uint32_t*)Acc, cnt);
  cnt = (size_t)n * 4;
  k_zero<<<(unsigned)((cnt + TB - 1) / TB), TB, 0, stream>>>((uint32_t*)Mbuf, cnt);
  k_zero<<<(unsigned)((cnt + TB - 1) / TB), TB, 0, stream>>>((uint32_t*)Den, cnt);

  cnt = (size_t)n * F;
  k_f32_to_f16<<<(unsigned)((cnt + TB - 1) / TB), TB, 0, stream>>>(xin, Xh, cnt);
  k_transpose_f16<<<(256 * F + TB - 1) / TB, TB, 0, stream>>>(W, WT, F);
  k_combine_att<<<(F * 8 + TB - 1) / TB, TB, 0, stream>>>(A, asrc, adst, Acomb, F);

  dim3 ggrid((n + 15) / 16, 4);
  k_wmma_gemm<<<ggrid, 128, 0, stream>>>(Xh, WT, Hbuf, n, F);

  k_alpha<<<(n * 8 + TB - 1) / TB, TB, 0, stream>>>(xin, Acomb, Alpha, n, F);

  unsigned eb = (unsigned)((E * 4 + TB - 1) / TB);
  k_edge_max<<<eb, TB, 0, stream>>>(ei, Alpha, Mbuf, E);
  k_edge_denom<<<eb, TB, 0, stream>>>(ei, Alpha, Mbuf, Den, E);
  k_edge_scatter<<<4096, TB, 0, stream>>>(ei, Alpha, Mbuf, Den, Hbuf, Acc, E);

  k_finalize<<<(n * 64 + TB - 1) / TB, TB, 0, stream>>>(Acc, bias, out, n);
}

extern "C" void kernel_launch(void* const* d_in, const int* in_sizes, int n_in,
                              void* d_out, int out_size, void* d_ws, size_t ws_size,
                              hipStream_t stream) {
  const float* x      = (const float*)d_in[0];
  const long long* ei = (const long long*)d_in[1];  // int64 [2, E]
  const float* W1     = (const float*)d_in[2];
  const float* A1     = (const float*)d_in[3];
  const float* asrc1  = (const float*)d_in[4];
  const float* adst1  = (const float*)d_in[5];
  const float* b1     = (const float*)d_in[6];
  const float* W2     = (const float*)d_in[7];
  const float* A2     = (const float*)d_in[8];
  const float* asrc2  = (const float*)d_in[9];
  const float* adst2  = (const float*)d_in[10];
  const float* b2     = (const float*)d_in[11];

  int F1 = in_sizes[2] / 256;           // 128
  int F2 = in_sizes[7] / 256;           // 64
  int n = in_sizes[0] / F1;             // 50000
  long long E = in_sizes[1] / 2;        // 800000

  char* ws = (char*)d_ws;
  size_t off = 0;
  auto alloc = [&](size_t bytes) -> char* {
    char* p = ws + off;
    off += (bytes + 255) & ~(size_t)255;
    return p;
  };
  float*    Hbuf  = (float*)alloc((size_t)n * 256 * sizeof(float));     // 51.2 MB
  float*    Acc   = (float*)alloc((size_t)n * 256 * sizeof(float));     // 51.2 MB
  float*    Alpha = (float*)alloc((size_t)n * 8 * sizeof(float));
  unsigned* Mbuf  = (unsigned*)alloc((size_t)n * 4 * sizeof(unsigned));
  float*    Den   = (float*)alloc((size_t)n * 4 * sizeof(float));
  _Float16* Xh    = (_Float16*)alloc((size_t)n * F1 * sizeof(_Float16));
  _Float16* WT    = (_Float16*)alloc((size_t)256 * F1 * sizeof(_Float16));
  float*    Acomb = (float*)alloc((size_t)F1 * 8 * sizeof(float));
  float* out = (float*)d_out;

  // Layer 1 writes its activations into d_out, which layer 2 consumes and overwrites.
  run_layer(x,   F1, n, E, ei, W1, A1, asrc1, adst1, b1, out,
            Hbuf, Acc, Alpha, Mbuf, Den, Xh, WT, Acomb, stream);
  run_layer(out, F2, n, E, ei, W2, A2, asrc2, adst2, b2, out,
            Hbuf, Acc, Alpha, Mbuf, Den, Xh, WT, Acomb, stream);
}
